// GRUv1_26379689132568
// MI455X (gfx1250) — compile-verified, matrix operand order checked by reading, not run on hardware
//
#include <hip/hip_runtime.h>

#define SEQ   256
#define BATCH 1024
#define IDIM  46
#define INDIM 40
#define HD    256
#define H3    768
#define BT    16

typedef __attribute__((ext_vector_type(8)))  unsigned short ushort8_t;
typedef __attribute__((ext_vector_type(16))) unsigned short ushort16_t;
typedef __attribute__((ext_vector_type(16))) __bf16         v16bf;
typedef __attribute__((ext_vector_type(8)))  float          v8f;

__device__ __forceinline__ unsigned short f2bf(float x) {
  unsigned int u = __builtin_bit_cast(unsigned int, x);
  u += 0x7FFFu + ((u >> 16) & 1u);           // round-to-nearest-even
  return (unsigned short)(u >> 16);
}

// 16-element bf16 fragment: two 16B chunks at p and p+16 elements
// (CDNA5 16-bit A/B VGPR layout: K k..k+7 then K k+16..k+23 per lane).
__device__ __forceinline__ v16bf ldfrag(const unsigned short* p) {
  ushort8_t a = *(const ushort8_t*)(p);
  ushort8_t b = *(const ushort8_t*)(p + 16);
  ushort16_t u = __builtin_shufflevector(a, b, 0,1,2,3,4,5,6,7,8,9,10,11,12,13,14,15);
  union { ushort16_t u; v16bf f; } cv; cv.u = u; return cv.f;
}

__device__ __forceinline__ v8f wmma_bf16(v16bf a, v16bf b, v8f c) {
  return __builtin_amdgcn_wmma_f32_16x16x32_bf16(false, a, false, b, (short)0, c, false, false);
}

// ---------------------------------------------------------------------------
// Stage 0: convert + K-pad weights to bf16 in workspace.
// Layout (elements): Whh[768][256] | Wih[768][64] | Wcomb[48][96] |
//                    W2a[256][64]  | W2b[256][256]
// ---------------------------------------------------------------------------
__global__ void prep_weights(const float* __restrict__ Whh_f, const float* __restrict__ Wih_f,
                             const float* __restrict__ Wcomb_f, const float* __restrict__ Wcom2_f,
                             unsigned short* __restrict__ W) {
  int i = blockIdx.x * blockDim.x + threadIdx.x;
  const int N0 = H3 * HD;
  const int N1 = H3 * 64;
  const int N2 = 48 * 96;
  const int N3 = HD * 64;
  const int N4 = HD * HD;
  if (i < N0) { W[i] = f2bf(Whh_f[i]); return; }
  i -= N0; unsigned short* P = W + N0;
  if (i < N1) { int n = i / 64, k = i % 64; P[i] = f2bf(k < 40 ? Wih_f[n * 40 + k] : 0.f); return; }
  i -= N1; P += N1;
  if (i < N2) { int n = i / 96, k = i % 96; P[i] = f2bf((n < 40 && k < 80) ? Wcomb_f[n * 80 + k] : 0.f); return; }
  i -= N2; P += N2;
  if (i < N3) { int n = i / 64, k = i % 64; P[i] = f2bf(k < 40 ? Wcom2_f[n * 296 + k] : 0.f); return; }
  i -= N3; P += N3;
  if (i < N4) { int n = i / 256, k = i % 256; P[i] = f2bf(Wcom2_f[n * 296 + 40 + k]); return; }
}

// ---------------------------------------------------------------------------
// Fused persistent scan: each workgroup owns 16 batch rows for all 256 steps.
// ---------------------------------------------------------------------------
__global__ __launch_bounds__(256) void gru_scan(
    const float* __restrict__ X, const float* __restrict__ Mm, const float* __restrict__ mask,
    const float* __restrict__ W_i2h, const float* __restrict__ b_i2h,
    const float* __restrict__ b_comb, const float* __restrict__ b_ih, const float* __restrict__ b_hh,
    const float* __restrict__ b_com2,
    const float* __restrict__ W_nn1, const float* __restrict__ b_nn1,
    const float* __restrict__ W_nn2, const float* __restrict__ b_nn2,
    const unsigned short* __restrict__ Whh, const unsigned short* __restrict__ Wih,
    const unsigned short* __restrict__ Wcb, const unsigned short* __restrict__ W2a,
    const unsigned short* __restrict__ W2b,
    float* __restrict__ out) {
  __shared__ float          hf[BT][HD];
  __shared__ unsigned short hb[BT][HD + 8];
  __shared__ unsigned short xmb[BT][96];
  __shared__ unsigned short cbuf[BT][64];
  __shared__ float          gs[BT][520];        // gx+gh, N 0..511 (r,z)
  __shared__ float          gnx[BT][260];       // gx, N 512..767
  __shared__ float          gnh[BT][260];       // gh, N 512..767
  __shared__ float          ot[BT][260];
  __shared__ float          o1s[BT][8];
  __shared__ float bsum[512];                   // b_ih + b_hh for r,z
  __shared__ float bihn[HD], bhhn[HD];          // n-gate biases
  __shared__ float bcp[48];                     // zero-padded b_comb
  __shared__ float bc2[HD];                     // b_com2
  __shared__ float wnn1s[8][HD];
  __shared__ float bnn1l[8], wnn2l[16], bnn2l[2];

  const int t    = threadIdx.x;
  const int wave = t >> 5;
  const int lane = t & 31;
  const int lr   = lane & 15;
  const int hi   = (lane >> 4) & 1;
  const int b0   = blockIdx.x * BT;

  // ---- init: LDS constants, cbuf pad zero, h0 ----
  for (int e = t; e < 512; e += 256) bsum[e] = b_ih[e] + b_hh[e];
  bihn[t] = b_ih[512 + t];
  bhhn[t] = b_hh[512 + t];
  bc2[t]  = b_com2[t];
  if (t < 48) bcp[t] = (t < 40) ? b_comb[t] : 0.f;
  for (int e = t; e < 8 * HD; e += 256) wnn1s[e >> 8][e & 255] = W_nn1[e];
  if (t < 8)  bnn1l[t] = b_nn1[t];
  if (t < 16) wnn2l[t] = W_nn2[t];
  if (t < 2)  bnn2l[t] = b_nn2[t];
  cbuf[t >> 4][48 + (t & 15)] = 0;
  {
    int r = t >> 4, c0 = (t & 15) * 16;
    for (int cc = 0; cc < 16; ++cc) {
      int c = c0 + cc;
      float h = b_i2h[c];
#pragma unroll
      for (int j = 0; j < 6; ++j)
        h += X[(0 * BATCH + b0 + r) * IDIM + INDIM + j] * W_i2h[c * 6 + j];
      hf[r][c] = h; hb[r][c] = f2bf(h);
    }
  }
  __syncthreads();

  for (int s = 0; s < SEQ; ++s) {
    // ---- phase 1: stage [x|m] tile as bf16 ----
    for (int e = t; e < BT * 96; e += 256) {
      int r = e / 96, k = e - r * 96;
      float v = 0.f;
      const int gbase = (s * BATCH + b0 + r) * IDIM;
      if (k < INDIM)          v = X[gbase + k];
      else if (k < 2 * INDIM) v = Mm[gbase + (k - INDIM)];
      xmb[r][k] = f2bf(v);
    }
    __syncthreads();

    // ---- phase 2: comb = xm @ Wcomb.T + bcp (branch-free store) ----
    if (wave < 3) {
      const int n0 = wave * 16;
      const unsigned short* pB = Wcb + (n0 + lr) * 96 + hi * 8;
      v16bf bF0 = ldfrag(pB);
      v16bf bF1 = ldfrag(pB + 32);
      v16bf bF2 = ldfrag(pB + 64);
      v16bf aF0 = ldfrag(&xmb[lr][0  + hi * 8]);
      v16bf aF1 = ldfrag(&xmb[lr][32 + hi * 8]);
      v16bf aF2 = ldfrag(&xmb[lr][64 + hi * 8]);
      v8f acc = {};
      acc = wmma_bf16(aF0, bF0, acc);
      acc = wmma_bf16(aF1, bF1, acc);
      acc = wmma_bf16(aF2, bF2, acc);
      const int n = n0 + lr;
      const float bias = bcp[n];
#pragma unroll
      for (int i = 0; i < 8; ++i) cbuf[i + 8 * hi][n] = f2bf(acc[i] + bias);
    }
    __syncthreads();

    // ---- phase 3: gx = comb@Wih.T ; gh = h@Whh.T  (8 waves x 6 N-tiles) ----
    // A fragments hoisted across tiles; B side uses a 2-deep ping-pong so each
    // load pair has ~2 WMMAs of latency cover without blowing registers.
    {
      v16bf aH[8];
#pragma unroll
      for (int k8 = 0; k8 < 8; ++k8) aH[k8] = ldfrag(&hb[lr][k8 * 32 + hi * 8]);
      v16bf aX0 = ldfrag(&cbuf[lr][0  + hi * 8]);
      v16bf aX1 = ldfrag(&cbuf[lr][32 + hi * 8]);
      for (int u = 0; u < 6; ++u) {
        const int nt = wave * 6 + u;
        const int n0 = nt * 16;
        const unsigned short* pX = Wih + (n0 + lr) * 64  + hi * 8;
        const unsigned short* pH = Whh + (n0 + lr) * 256 + hi * 8;
        v8f accX = {};
        {
          v16bf bX0 = ldfrag(pX);
          v16bf bX1 = ldfrag(pX + 32);
          accX = wmma_bf16(aX0, bX0, accX);
          accX = wmma_bf16(aX1, bX1, accX);
        }
        v8f accH = {};
        v16bf c0 = ldfrag(pH);
        v16bf c1 = ldfrag(pH + 32);
#pragma unroll
        for (int kp = 0; kp < 4; ++kp) {
          v16bf n0f = c0, n1f = c1;
          if (kp < 3) { n0f = ldfrag(pH + (2 * kp + 2) * 32); n1f = ldfrag(pH + (2 * kp + 3) * 32); }
          accH = wmma_bf16(aH[2 * kp],     c0, accH);
          accH = wmma_bf16(aH[2 * kp + 1], c1, accH);
          c0 = n0f; c1 = n1f;
        }
        if (nt < 32) {
#pragma unroll
          for (int i = 0; i < 8; ++i) gs[i + 8 * hi][n0 + lr] = accX[i] + accH[i];
        } else {
          const int c = n0 - 512 + lr;
#pragma unroll
          for (int i = 0; i < 8; ++i) { gnx[i + 8 * hi][c] = accX[i]; gnh[i + 8 * hi][c] = accH[i]; }
        }
      }
    }
    __syncthreads();

    // ---- phase 4: gates + masked h update (rolled loop, scalar stores) ----
    {
      const int r = t >> 4, c0 = (t & 15) * 16;
      const float mv = mask[s * BATCH + b0 + r];
      for (int cc = 0; cc < 16; ++cc) {
        int c = c0 + cc;
        float hprev = hf[r][c];
        float gr = gs[r][c]       + bsum[c];
        float gz = gs[r][256 + c] + bsum[256 + c];
        float rg = 1.f / (1.f + expf(-gr));
        float zg = 1.f / (1.f + expf(-gz));
        float ng = tanhf((gnx[r][c] + bihn[c]) + rg * (gnh[r][c] + bhhn[c]));
        float hn = (1.f - zg) * ng + zg * hprev;
        float hx = hn * mv + hprev * (1.f - mv);
        hf[r][c] = hx; hb[r][c] = f2bf(hx);
      }
    }
    __syncthreads();

    // ---- phase 5: out = comb@W2a.T + h@W2b.T + b_com2 (8 waves x 2 tiles) ----
    {
      v16bf aH[8];
#pragma unroll
      for (int k8 = 0; k8 < 8; ++k8) aH[k8] = ldfrag(&hb[lr][k8 * 32 + hi * 8]);
      v16bf aX0 = ldfrag(&cbuf[lr][0  + hi * 8]);
      v16bf aX1 = ldfrag(&cbuf[lr][32 + hi * 8]);
      for (int u = 0; u < 2; ++u) {
        const int n0 = (wave * 2 + u) * 16;
        const unsigned short* pA = W2a + (n0 + lr) * 64  + hi * 8;
        const unsigned short* pB = W2b + (n0 + lr) * 256 + hi * 8;
        v8f acc = {};
        {
          v16bf bX0 = ldfrag(pA);
          v16bf bX1 = ldfrag(pA + 32);
          acc = wmma_bf16(aX0, bX0, acc);
          acc = wmma_bf16(aX1, bX1, acc);
        }
        v16bf c0 = ldfrag(pB);
        v16bf c1 = ldfrag(pB + 32);
#pragma unroll
        for (int kp = 0; kp < 4; ++kp) {
          v16bf n0f = c0, n1f = c1;
          if (kp < 3) { n0f = ldfrag(pB + (2 * kp + 2) * 32); n1f = ldfrag(pB + (2 * kp + 3) * 32); }
          acc = wmma_bf16(aH[2 * kp],     c0, acc);
          acc = wmma_bf16(aH[2 * kp + 1], c1, acc);
          c0 = n0f; c1 = n1f;
        }
        const float bias = bc2[n0 + lr];
#pragma unroll
        for (int i = 0; i < 8; ++i) ot[i + 8 * hi][n0 + lr] = acc[i] + bias;
      }
    }
    __syncthreads();

    // ---- phase 6: o1 = relu(ot @ Wnn1.T + b1), weights from LDS ----
    if (t < 128) {
      const int r = t >> 3, j = t & 7;
      float sacc = bnn1l[j];
      for (int c = 0; c < HD; ++c) sacc += wnn1s[j][c] * ot[r][c];
      o1s[r][j] = sacc > 0.f ? sacc : 0.f;
    }
    __syncthreads();

    // ---- phase 7: final 2-wide head -> d_out ----
    if (t < 32) {
      const int r = t >> 1, o = t & 1;
      float sacc = bnn2l[o];
#pragma unroll
      for (int j = 0; j < 8; ++j) sacc += wnn2l[o * 8 + j] * o1s[r][j];
      out[(s * BATCH + b0 + r) * 2 + o] = sacc;
    }
    __syncthreads();
  }
}

extern "C" void kernel_launch(void* const* d_in, const int* in_sizes, int n_in,
                              void* d_out, int out_size, void* d_ws, size_t ws_size,
                              hipStream_t stream) {
  const float* X      = (const float*)d_in[0];
  const float* Mm     = (const float*)d_in[1];
  const float* mask   = (const float*)d_in[2];
  const float* W_i2h  = (const float*)d_in[3];
  const float* b_i2h  = (const float*)d_in[4];
  const float* W_comb = (const float*)d_in[5];
  const float* b_comb = (const float*)d_in[6];
  const float* W_ih   = (const float*)d_in[7];
  const float* W_hh   = (const float*)d_in[8];
  const float* b_ih   = (const float*)d_in[9];
  const float* b_hh   = (const float*)d_in[10];
  const float* W_com2 = (const float*)d_in[11];
  const float* b_com2 = (const float*)d_in[12];
  const float* W_nn1  = (const float*)d_in[13];
  const float* b_nn1  = (const float*)d_in[14];
  const float* W_nn2  = (const float*)d_in[15];
  const float* b_nn2  = (const float*)d_in[16];

  unsigned short* W = (unsigned short*)d_ws;
  const int total = H3 * HD + H3 * 64 + 48 * 96 + HD * 64 + HD * HD;  // 332288
  prep_weights<<<(total + 255) / 256, 256, 0, stream>>>(W_hh, W_ih, W_comb, W_com2, W);

  const unsigned short* Whh = W;
  const unsigned short* Wih = Whh + H3 * HD;
  const unsigned short* Wcb = Wih + H3 * 64;
  const unsigned short* W2a = Wcb + 48 * 96;
  const unsigned short* W2b = W2a + HD * 64;

  gru_scan<<<BATCH / BT, 256, 0, stream>>>(
      X, Mm, mask, W_i2h, b_i2h, b_comb, b_ih, b_hh, b_com2,
      W_nn1, b_nn1, W_nn2, b_nn2, Whh, Wih, Wcb, W2a, W2b, (float*)d_out);
}
